// RITS_36756330119285
// MI455X (gfx1250) — matrix operand order repaired and checked by reading, not compiled, not run on hardware
//
#include <hip/hip_runtime.h>
#include <hip/hip_bf16.h>
#include <math.h>

typedef __attribute__((ext_vector_type(16))) __bf16 v16bf;
typedef __attribute__((ext_vector_type(8)))  float  v8f;

#define RB 8192
#define RT 223
#define RD 7
#define RH 128
#define RE 50
#define RO 29
#define RA 9
#define KPAD 160     // 14 (cc,m) + 2 pad + 128 (h) + 16 pad
#define NG  512      // 4*H gate outputs
#define ROWS 16      // batch rows per workgroup
#define NTHR 512     // 16 waves

// ---------------- LDS byte layout (dynamic shared) ----------------
#define OFF_IN    0                          // ushort [ROWS*KPAD] bf16 activations
#define OFF_GATE  (OFF_IN + ROWS*KPAD*2)     // float  [ROWS*NG]
#define OFF_H     (OFF_GATE + ROWS*NG*4)     // float  [ROWS*RH]
#define OFF_C     (OFF_H + ROWS*RH*4)        // float  [ROWS*RH]
#define OFF_WTDH  (OFF_C + ROWS*RH*4)        // float  [RH*RD]
#define OFF_BTDH  (OFF_WTDH + RH*RD*4)       // float  [RH]
#define OFF_WHIST (OFF_BTDH + RH*4)          // float  [RD*RH]
#define OFF_BG    (OFF_WHIST + RD*RH*4)      // float  [NG]  (b_ih + b_hh)
#define OFF_X     (OFF_BG + NG*4)            // float  [112] each
#define OFF_M     (OFF_X + 112*4)
#define OFF_DEL   (OFF_M + 112*4)
#define OFF_XH    (OFF_DEL + 112*4)
#define OFF_XC    (OFF_XH + 112*4)
#define OFF_GX    (OFF_XC + 112*4)
#define OFF_CC    (OFF_GX + 112*4)
#define OFF_WF    (OFF_CC + 112*4)           // float [49] pad->52
#define OFF_WCOMB (OFF_WF + 52*4)            // float [98] pad->100
#define OFF_BHIST (OFF_WCOMB + 100*4)        // 8 floats each below
#define OFF_BFEAT (OFF_BHIST + 8*4)
#define OFF_BCOMB (OFF_BFEAT + 8*4)
#define OFF_TDD   (OFF_BCOMB + 8*4)
#define OFF_BTDX  (OFF_TDD + 8*4)
#define OFF_RED   (OFF_BTDX + 8*4)           // 4 floats: n1,n2,n3,msum partials
#define SMEM_TOTAL (OFF_RED + 4*4)           // ~66 KB -> multiple WGs per WGP

union Frag  { uint4 q[2];            v16bf v; };
union FragU { unsigned short s[16];  v16bf v; };

__device__ __forceinline__ unsigned short f32_to_bf16_rne(float f) {
    unsigned int u = __float_as_uint(f);
    unsigned int lsb = (u >> 16) & 1u;
    u += 0x7fffu + lsb;
    return (unsigned short)(u >> 16);
}
__device__ __forceinline__ float sigmf(float x) { return 1.0f / (1.0f + __expf(-x)); }

// fused K-padded gate weight: k<14 -> W_ih, 16..143 -> W_hh, else 0
__device__ __forceinline__ float fusedW(const float* __restrict__ W_ih,
                                        const float* __restrict__ W_hh, int n, int k) {
    if (k < 14)               return W_ih[n * 14 + k];
    if (k >= 16 && k < 144)   return W_hh[n * RH + (k - 16)];
    return 0.0f;
}

__global__ __launch_bounds__(NTHR)
void rits_main_kernel(const float* __restrict__ values, const float* __restrict__ masks,
                      const float* __restrict__ deltas, const float* __restrict__ probs,
                      const float* __restrict__ anc_in, const int* __restrict__ labels,
                      const float* __restrict__ W_td_h, const float* __restrict__ b_td_h,
                      const float* __restrict__ W_td_x, const float* __restrict__ b_td_x,
                      const float* __restrict__ W_hist, const float* __restrict__ b_hist,
                      const float* __restrict__ W_feat, const float* __restrict__ b_feat,
                      const float* __restrict__ W_comb, const float* __restrict__ b_comb,
                      const float* __restrict__ W_ih,   const float* __restrict__ W_hh,
                      const float* __restrict__ b_ih,   const float* __restrict__ b_hh,
                      const float* __restrict__ W_anc,  const float* __restrict__ b_anc,
                      const float* __restrict__ W_cat,  const float* __restrict__ b_cat,
                      const float* __restrict__ W_out,  const float* __restrict__ b_out,
                      float* __restrict__ out_yh, float* __restrict__ out_imp,
                      float* __restrict__ out_lab,
                      float* __restrict__ ws_msum, float* __restrict__ ws_n1,
                      float* __restrict__ ws_n2,   float* __restrict__ ws_n3,
                      float* __restrict__ ws_ynum)
{
    extern __shared__ char smem[];
    unsigned short* sIn = (unsigned short*)(smem + OFF_IN);
    float* sGate  = (float*)(smem + OFF_GATE);
    float* sH     = (float*)(smem + OFF_H);
    float* sC     = (float*)(smem + OFF_C);
    float* sWtdh  = (float*)(smem + OFF_WTDH);
    float* sBtdh  = (float*)(smem + OFF_BTDH);
    float* sWhist = (float*)(smem + OFF_WHIST);
    float* sBg    = (float*)(smem + OFF_BG);
    float* sX     = (float*)(smem + OFF_X);
    float* sM     = (float*)(smem + OFF_M);
    float* sDel   = (float*)(smem + OFF_DEL);
    float* sXh    = (float*)(smem + OFF_XH);
    float* sXc    = (float*)(smem + OFF_XC);
    float* sGx    = (float*)(smem + OFF_GX);
    float* sCc    = (float*)(smem + OFF_CC);
    float* sWf    = (float*)(smem + OFF_WF);
    float* sWcomb = (float*)(smem + OFF_WCOMB);
    float* sBhist = (float*)(smem + OFF_BHIST);
    float* sBfeat = (float*)(smem + OFF_BFEAT);
    float* sBcomb = (float*)(smem + OFF_BCOMB);
    float* sTdd   = (float*)(smem + OFF_TDD);
    float* sBtdx  = (float*)(smem + OFF_BTDX);
    float* sRed   = (float*)(smem + OFF_RED);

    const int tid    = threadIdx.x;
    const int row    = tid >> 5;       // 0..15  (batch row in tile; also wave id)
    const int lane32 = tid & 31;       // lane in wave
    const int wv     = tid >> 5;       // 0..15  wave id (2 n-tiles each)
    const int lrow   = lane32 & 15;
    const bool hihalf = lane32 >= 16;
    const long long brow = (long long)blockIdx.x * ROWS + row;

    // ---- one-time: small weights + biases into LDS ----
    for (int e = tid; e < RH * RD; e += NTHR) sWtdh[e] = W_td_h[e];
    for (int e = tid; e < RH; e += NTHR) sBtdh[e] = b_td_h[e];
    for (int e = tid; e < RD * RH; e += NTHR) sWhist[e] = W_hist[e];
    for (int e = tid; e < NG; e += NTHR) sBg[e] = b_ih[e] + b_hh[e];
    for (int e = tid; e < 49; e += NTHR) { int r = e / 7, c2 = e - r * 7; sWf[e] = (r == c2) ? 0.0f : W_feat[e]; }
    for (int e = tid; e < 98; e += NTHR) sWcomb[e] = W_comb[e];
    if (tid < RD) {
        sBhist[tid] = b_hist[tid];
        sBfeat[tid] = b_feat[tid];
        sBcomb[tid] = b_comb[tid];
        sTdd[tid]   = W_td_x[tid * RD + tid];   // diagonal decay
        sBtdx[tid]  = b_td_x[tid];
    }
    for (int e = tid; e < ROWS * RH; e += NTHR) { sH[e] = 0.0f; sC[e] = 0.0f; }
    for (int e = tid; e < ROWS * KPAD; e += NTHR) sIn[e] = 0;   // zeros incl. K pads

    // ---- one-time: register-resident B fragments (per-wave weight columns) ----
    // Wave wv owns gate columns [(wv*2)*16, (wv*2+2)*16). K order used by this code (same as A):
    // lane holds 16 contiguous K at kw = ks*32 + (hi?16:0) for column n = tile*16 + lrow.
    v16bf Bfrag[2][5];
    #pragma unroll
    for (int nt = 0; nt < 2; ++nt) {
        const int n = (wv * 2 + nt) * 16 + lrow;
        #pragma unroll
        for (int ks = 0; ks < 5; ++ks) {
            const int kw = ks * 32 + (hihalf ? 16 : 0);
            FragU fu;
            #pragma unroll
            for (int i = 0; i < 16; ++i)
                fu.s[i] = f32_to_bf16_rne(fusedW(W_ih, W_hh, n, kw + i));
            Bfrag[nt][ks] = fu.v;
        }
    }
    __syncthreads();

    // =================== sequence loop ===================
    for (int t = 0; t < RT; ++t) {
        // load x,m,d for the 16 rows at time t
        for (int e = tid; e < ROWS * RD; e += NTHR) {
            int r = e / RD, dc = e - r * RD;
            long long off = ((long long)(blockIdx.x * ROWS + r) * RT + t) * RD + dc;
            sX[e]   = values[off];
            sM[e]   = masks[off];
            sDel[e] = deltas[off];
        }
        if (tid < 4) sRed[tid] = 0.0f;
        __syncthreads();

        // h *= gamma_h  (each thread: 4 hidden units of its row)
        {
            const float* drow = &sDel[row * RD];
            #pragma unroll
            for (int q = 0; q < 4; ++q) {
                int j = lane32 * 4 + q;
                float acc = sBtdh[j];
                #pragma unroll
                for (int k = 0; k < RD; ++k) acc += drow[k] * sWtdh[j * RD + k];
                sH[row * RH + j] *= __expf(-fmaxf(acc, 0.0f));
            }
        }
        __syncthreads();

        // x_h, gamma_x, x_c, loss term 1 + msum
        if (lane32 < RD) {
            int dc = lane32;
            float acc = sBhist[dc];
            const float* hr = &sH[row * RH];
            #pragma unroll 8
            for (int k = 0; k < RH; ++k) acc += hr[k] * sWhist[dc * RH + k];
            sXh[row * RD + dc] = acc;
            float x = sX[row * RD + dc], m = sM[row * RD + dc];
            float gx = __expf(-fmaxf(sDel[row * RD + dc] * sTdd[dc] + sBtdx[dc], 0.0f));
            sGx[row * RD + dc] = gx;
            sXc[row * RD + dc] = m * x + (1.0f - m) * acc;
            atomicAdd(&sRed[3], m);
            atomicAdd(&sRed[0], fabsf(x - acc) * m);
        }
        __syncthreads();

        // z_h, alpha, c_h, c_c, loss terms 2 & 3, imputation out
        if (lane32 < RD) {
            int dc = lane32;
            float zh = sBfeat[dc];
            #pragma unroll
            for (int k = 0; k < RD; ++k) zh += sXc[row * RD + k] * sWf[dc * RD + k];
            float alpha = sBcomb[dc];
            #pragma unroll
            for (int k = 0; k < RD; ++k) alpha += sGx[row * RD + k] * sWcomb[dc * 14 + k];
            #pragma unroll
            for (int k = 0; k < RD; ++k) alpha += sM[row * RD + k] * sWcomb[dc * 14 + 7 + k];
            float xh = sXh[row * RD + dc];
            float ch = alpha * zh + (1.0f - alpha) * xh;
            float x = sX[row * RD + dc], m = sM[row * RD + dc];
            float ccv = m * x + (1.0f - m) * ch;
            atomicAdd(&sRed[1], fabsf(x - zh) * m);
            atomicAdd(&sRed[2], fabsf(x - ch) * m);
            sCc[row * RD + dc] = ccv;
            out_imp[(brow * RT + t) * RD + dc] = ccv;
        }
        __syncthreads();

        // build bf16 activation tile [row][k<144]: cc(7) | m(7) | 0 0 | h(128); 144.. stays 0
        for (int e = tid; e < ROWS * 144; e += NTHR) {
            int r = e / 144, k = e - r * 144;
            float v = 0.0f;
            if (k < 7)       v = sCc[r * RD + k];
            else if (k < 14) v = sM[r * RD + (k - 7)];
            else if (k >= 16) v = sH[r * RH + (k - 16)];
            sIn[r * KPAD + k] = f32_to_bf16_rne(v);
        }
        // flush per-t loss partials (threads 508..511)
        if (tid >= NTHR - 4) {
            int q = tid - (NTHR - 4);
            float v = sRed[q];
            float* tgt = (q == 0) ? &ws_n1[t] : (q == 1) ? &ws_n2[t]
                       : (q == 2) ? &ws_n3[t] : &ws_msum[t];
            atomicAdd(tgt, v);
        }
        __syncthreads();

        // ---------- WMMA GEMM: gates[16][512] = In[16][160] @ Wt[512][160]^T ----------
        // B register-resident; prefetch ALL A fragments first so the 10 ds_load_b128
        // pipeline against the WMMA chain (incremental s_wait_dscnt instead of 5 stalls).
        {
            const int ca_base = hihalf ? 16 : 0;   // A K-offset per lane half
            Frag fa[5];
            #pragma unroll
            for (int ks = 0; ks < 5; ++ks) {
                const int kb = ks * 32;
                fa[ks].q[0] = *(const uint4*)&sIn[lrow * KPAD + kb + ca_base];
                fa[ks].q[1] = *(const uint4*)&sIn[lrow * KPAD + kb + ca_base + 8];
            }
            v8f acc0 = {}, acc1 = {};
            #pragma unroll
            for (int ks = 0; ks < 5; ++ks) {
                acc0 = __builtin_amdgcn_wmma_f32_16x16x32_bf16(
                    false, fa[ks].v, false, Bfrag[0][ks], (short)0, acc0, false, false);
                acc1 = __builtin_amdgcn_wmma_f32_16x16x32_bf16(
                    false, fa[ks].v, false, Bfrag[1][ks], (short)0, acc1, false, false);
            }
            const int mb = hihalf ? 8 : 0;
            const int n0 = (wv * 2 + 0) * 16 + lrow;
            const int n1 = (wv * 2 + 1) * 16 + lrow;
            #pragma unroll
            for (int r = 0; r < 8; ++r) {
                sGate[(mb + r) * NG + n0] = acc0[r];
                sGate[(mb + r) * NG + n1] = acc1[r];
            }
        }
        __syncthreads();

        // ---------- LSTM pointwise update (f32 state) ----------
        {
            #pragma unroll
            for (int q = 0; q < 4; ++q) {
                int j = lane32 * 4 + q;
                float gi = sGate[row * NG + j]            + sBg[j];
                float gf = sGate[row * NG + RH + j]       + sBg[RH + j];
                float gg = sGate[row * NG + 2 * RH + j]   + sBg[2 * RH + j];
                float go = sGate[row * NG + 3 * RH + j]   + sBg[3 * RH + j];
                float cn = sigmf(gf) * sC[row * RH + j] + sigmf(gi) * tanhf(gg);
                sC[row * RH + j] = cn;
                sH[row * RH + j] = sigmf(go) * tanhf(cn);
            }
        }
        __syncthreads();
    }

    // =================== final head ===================
    float* sAnc = sGate;                 // reuse: [16][50]
    float* sH2  = &sGate[ROWS * 64];     // reuse: [16][128]
    for (int e = tid; e < ROWS * RE; e += NTHR) {
        int r = e / RE, u = e - r * RE;
        float acc = b_anc[u];
        const float* arow = &anc_in[(long long)(blockIdx.x * ROWS + r) * RA];
        #pragma unroll
        for (int k = 0; k < RA; ++k) acc += arow[k] * W_anc[u * RA + k];
        sAnc[e] = fmaxf(acc, 0.0f);
    }
    __syncthreads();
    #pragma unroll
    for (int q = 0; q < 4; ++q) {
        int j = lane32 * 4 + q;
        float acc = b_cat[j];
        const float* wr = &W_cat[j * (RH + RE)];
        #pragma unroll 8
        for (int k = 0; k < RH; ++k) acc += sH[row * RH + k] * wr[k];
        for (int k = 0; k < RE; ++k) acc += sAnc[row * RE + k] * wr[RH + k];
        sH2[row * RH + j] = fmaxf(acc, 0.0f);
    }
    __syncthreads();
    if (lane32 == 0) {
        float logit[RO];
        float mx = -1e30f;
        for (int o = 0; o < RO; ++o) {
            float acc = b_out[o];
            for (int k = 0; k < RH; ++k) acc += sH2[row * RH + k] * W_out[o * RH + k];
            logit[o] = acc;
            mx = fmaxf(mx, acc);
        }
        float s = 0.0f;
        for (int o = 0; o < RO; ++o) { logit[o] = __expf(logit[o] - mx); s += logit[o]; }
        float inv = 1.0f / s, yl = 0.0f;
        for (int o = 0; o < RO; ++o) {
            float y = logit[o] * inv;
            out_yh[brow * RO + o] = y;
            float dd = y - probs[brow * RO + o];
            yl += dd * dd;
        }
        atomicAdd(ws_ynum, yl);
        out_lab[brow] = (float)labels[brow];
    }
}

__global__ void rits_zero_ws(float* ws, int n) {
    int i = blockIdx.x * blockDim.x + threadIdx.x;
    if (i < n) ws[i] = 0.0f;
}

__global__ void rits_finalize(const float* __restrict__ ws_msum, const float* __restrict__ n1,
                              const float* __restrict__ n2, const float* __restrict__ n3,
                              const float* __restrict__ ynum, float* __restrict__ out) {
    if (threadIdx.x == 0 && blockIdx.x == 0) {
        float xl = 0.0f;
        for (int t = 0; t < RT; ++t)
            xl += (n1[t] + n2[t] + n3[t]) / (ws_msum[t] + 1e-5f);
        xl *= 0.3f;                               // IMPUTE_W
        float yl = ynum[0] / ((float)RB + 1e-5f); // LABEL_W = 1.0
        out[0] = xl;
        out[1] = yl;
        out[2] = xl + yl;
    }
}

extern "C" void kernel_launch(void* const* d_in, const int* in_sizes, int n_in,
                              void* d_out, int out_size, void* d_ws, size_t ws_size,
                              hipStream_t stream) {
    const float* values  = (const float*)d_in[0];
    const float* masks   = (const float*)d_in[1];
    const float* deltas  = (const float*)d_in[2];
    const float* probs   = (const float*)d_in[3];
    const float* anc_in  = (const float*)d_in[4];
    const int*   labels  = (const int*)  d_in[5];
    const float* W_td_h  = (const float*)d_in[6];
    const float* b_td_h  = (const float*)d_in[7];
    const float* W_td_x  = (const float*)d_in[8];
    const float* b_td_x  = (const float*)d_in[9];
    const float* W_hist  = (const float*)d_in[10];
    const float* b_hist  = (const float*)d_in[11];
    const float* W_feat  = (const float*)d_in[12];
    const float* b_feat  = (const float*)d_in[13];
    const float* W_comb  = (const float*)d_in[14];
    const float* b_comb  = (const float*)d_in[15];
    const float* W_ih    = (const float*)d_in[16];
    const float* W_hh    = (const float*)d_in[17];
    const float* b_ih    = (const float*)d_in[18];
    const float* b_hh    = (const float*)d_in[19];
    const float* W_anc   = (const float*)d_in[20];
    const float* b_anc   = (const float*)d_in[21];
    const float* W_cat   = (const float*)d_in[22];
    const float* b_cat   = (const float*)d_in[23];
    const float* W_out   = (const float*)d_in[24];
    const float* b_out   = (const float*)d_in[25];

    float* out = (float*)d_out;
    float* out_losses = out;                               // [3]
    float* out_yh     = out + 3;                           // [RB*RO]
    float* out_imp    = out_yh + (long long)RB * RO;       // [RB*RT*RD]
    float* out_lab    = out_imp + (long long)RB * RT * RD; // [RB]

    float* ws      = (float*)d_ws;
    float* ws_msum = ws;          // [256]
    float* ws_n1   = ws + 256;    // [256]
    float* ws_n2   = ws + 512;    // [256]
    float* ws_n3   = ws + 768;    // [256]
    float* ws_ynum = ws + 1024;   // [1]

    (void)hipFuncSetAttribute((const void*)rits_main_kernel,
                              hipFuncAttributeMaxDynamicSharedMemorySize, SMEM_TOTAL);

    rits_zero_ws<<<5, 256, 0, stream>>>(ws, 1025);
    rits_main_kernel<<<RB / ROWS, NTHR, SMEM_TOTAL, stream>>>(
        values, masks, deltas, probs, anc_in, labels,
        W_td_h, b_td_h, W_td_x, b_td_x, W_hist, b_hist, W_feat, b_feat,
        W_comb, b_comb, W_ih, W_hh, b_ih, b_hh, W_anc, b_anc, W_cat, b_cat,
        W_out, b_out,
        out_yh, out_imp, out_lab, ws_msum, ws_n1, ws_n2, ws_n3, ws_ynum);
    rits_finalize<<<1, 64, 0, stream>>>(ws_msum, ws_n1, ws_n2, ws_n3, ws_ynum, out_losses);
    (void)out_losses; (void)out_size; (void)ws_size; (void)n_in; (void)in_sizes;
}
